// new_71476845740258
// MI455X (gfx1250) — matrix operand order repaired and checked by reading, not compiled, not run on hardware
//
#include <hip/hip_runtime.h>
#include <hip/hip_bf16.h>
#include <math.h>

typedef __attribute__((ext_vector_type(16))) __bf16          v16bf;
typedef __attribute__((ext_vector_type(16))) unsigned short  v16us;
typedef __attribute__((ext_vector_type(8)))  float           v8f;

#define FEAT   620
#define NH     128
#define STATE  2176
#define NH3    256
#define BSEQ   8
#define TSEQ   2048
#define MTOT   (BSEQ * TSEQ)   /* 16384 */
#define KC     20              /* k-chunks of 32 covering 640 (620 padded) */
#define NT     8               /* 8 n-tiles of 16 -> 128 */
#define NBLK   32              /* recurrence blocks */
#define RTH    128             /* recurrence threads per block */

/* ---- workspace layout (bytes) ---- */
#define OFF_P       0u                          /* 16384*128 f32 = 8 MB   */
#define OFF_WPK     (8u * 1024u * 1024u)        /* 8*20*32*8 u32 = 160 KB */
#define OFF_W0S     (OFF_WPK + 163840u)         /* 128*1088 u32           */
#define OFF_W1      (OFF_W0S + 557056u)         /* 128*64 u32             */
#define OFF_W2      (OFF_W1  + 32768u)          /* 2176*64 u32            */
#define OFF_GSTATE  (OFF_W2  + 557056u)         /* 2176 f32               */
#define OFF_GH0     (OFF_GSTATE + 8704u)        /* 128 f32                */
#define OFF_GH1     (OFF_GH0 + 512u)            /* 128 f32                */
#define OFF_SYNC    (OFF_GH1 + 512u)            /* 2 u32                  */

/* ---- bf16 helpers (round-to-nearest-even) ---- */
static __device__ __forceinline__ unsigned short f2bf(float f) {
    unsigned u = __builtin_bit_cast(unsigned, f);
    u += 0x7fffu + ((u >> 16) & 1u);
    return (unsigned short)(u >> 16);
}
static __device__ __forceinline__ unsigned pack2(float a, float b) {
    return (unsigned)f2bf(a) | ((unsigned)f2bf(b) << 16);
}
static __device__ __forceinline__ float bflo(unsigned u) {
    return __builtin_bit_cast(float, u << 16);
}
static __device__ __forceinline__ float bfhi(unsigned u) {
    return __builtin_bit_cast(float, u & 0xffff0000u);
}

static __device__ __forceinline__ float wred(float v) {
#pragma unroll
    for (int o = 16; o > 0; o >>= 1) v += __shfl_down(v, o, 32);
    return v;
}

/* device-wide sense-reversing barrier (L2 atomics) */
static __device__ __forceinline__ void gsync(unsigned* cnt, unsigned* gen, unsigned nblk) {
    __syncthreads();
    if (threadIdx.x == 0) {
        __threadfence();
        volatile unsigned* vgen = (volatile unsigned*)gen;
        unsigned g = *vgen;
        unsigned arrived = atomicAdd(cnt, 1u) + 1u;
        if (arrived == nblk) {
            atomicExch(cnt, 0u);
            __threadfence();
            atomicAdd(gen, 1u);
        } else {
            while (*vgen == g) { __builtin_amdgcn_s_sleep(1); }
        }
        __threadfence();
    }
    __syncthreads();
}

/* ---- kernel 0: init state + sync flags ---- */
__global__ void rnn_init(float* gstate, unsigned* sync2) {
    int i = blockIdx.x * blockDim.x + threadIdx.x;
    if (i < STATE) gstate[i] = 0.f;
    if (i < 2) sync2[i] = 0u;
}

/* ---- kernel 1: pack weights to bf16 (WMMA B-frag layout + pair-packed rows) ---- */
__global__ __launch_bounds__(256) void rnn_pack(
    const float* __restrict__ Wh0, const float* __restrict__ Wh1,
    const float* __restrict__ Wh2,
    unsigned* __restrict__ wpk, unsigned* __restrict__ w0s,
    unsigned* __restrict__ w1, unsigned* __restrict__ w2)
{
    int idx = blockIdx.x * 256 + threadIdx.x;
    if (idx < 40960) {
        /* WMMA B fragments for Wh0x^T: [nt][kc][lane][8 u32]
           lanes 0-15: K = kc*32 + 2v,2v+1 ; lanes 16-31: K = kc*32+16+2v ; N = lane%16 */
        int v = idx & 7, lane = (idx >> 3) & 31, rem = idx >> 8;
        int kc = rem % KC, nt = rem / KC;
        int n = nt * 16 + (lane & 15), half = (lane >> 4) & 1;
        int k = kc * 32 + half * 16 + 2 * v;
        float f0 = 0.f, f1 = 0.f;
        if (k < FEAT) { f0 = Wh0[(size_t)n * (FEAT + STATE) + k];
                        f1 = Wh0[(size_t)n * (FEAT + STATE) + k + 1]; }
        wpk[idx] = pack2(f0, f1);
    } else if (idx < 40960 + 139264) {
        int i = idx - 40960;                     /* Wh0s: [128][1088] pairs */
        int n = i / 1088, k2 = i % 1088;
        const float* r = Wh0 + (size_t)n * (FEAT + STATE) + FEAT + 2 * k2;
        w0s[i] = pack2(r[0], r[1]);
    } else if (idx < 40960 + 139264 + 8192) {
        int i = idx - (40960 + 139264);          /* Wh1: [128][64] pairs */
        int n = i / 64, k2 = i % 64;
        const float* r = Wh1 + (size_t)n * NH + 2 * k2;
        w1[i] = pack2(r[0], r[1]);
    } else if (idx < 40960 + 139264 + 8192 + 139264) {
        int i = idx - (40960 + 139264 + 8192);   /* Wh2: [2176][64] pairs */
        int n = i / 64, k2 = i % 64;
        const float* r = Wh2 + (size_t)n * NH + 2 * k2;
        w2[i] = pack2(r[0], r[1]);
    }
}

/* ---- kernel 2: P = X @ Wh0x^T + bh0, WMMA bf16 ----
   grid: 128 blocks * 8 waves = 1024 row-tiles of 16; each wave -> 16x128 tile.
   All 8 B fragments are loaded before the WMMA chain so the 16 b128 loads
   issue as one clause and overlap with the A-fragment convert VALU work. */
__global__ __launch_bounds__(256) void rnn_gemm(
    const float* __restrict__ fvs, const float* __restrict__ bh0,
    const unsigned* __restrict__ wpk, float* __restrict__ P)
{
    const int wave = threadIdx.x >> 5, lane = threadIdx.x & 31;
    const int rt = blockIdx.x * 8 + wave;
    const int half = (lane >> 4) & 1;
    const int m0 = rt * 16;
    const float* xrow = fvs + (size_t)(m0 + (lane & 15)) * FEAT;

    v8f acc[NT] = {};

    /* main chunks kc = 0..18: every k in range, no predication */
    for (int kc = 0; kc < KC - 1; ++kc) {
        const int kbase = kc * 32;
        /* stage all 8 B fragments (one clause of b128 loads) */
        v16bf b[NT];
#pragma unroll
        for (int nt = 0; nt < NT; ++nt)
            b[nt] = *(const v16bf*)(wpk + (((nt * KC + kc) * 32 + lane) << 3));
        /* prefetch next chunk's A data while loads are in flight */
        __builtin_prefetch(xrow + kbase + 32, 0, 3);
        /* A fragment: 16x32 bf16, contiguous k = 16*half + s per lane (convert overlaps loads) */
        v16us au{};
#pragma unroll
        for (int v = 0; v < 8; ++v) {
            int k = kbase + (half << 4) + (v << 1);
            float2 xy = *(const float2*)(xrow + k);
            au[2 * v]     = f2bf(xy.x);
            au[2 * v + 1] = f2bf(xy.y);
        }
        v16bf a = __builtin_bit_cast(v16bf, au);
#pragma unroll
        for (int nt = 0; nt < NT; ++nt)
            acc[nt] = __builtin_amdgcn_wmma_f32_16x16x32_bf16(
                false, a, false, b[nt], (short)0, acc[nt], false, false);
    }

    /* tail chunk kc = 19: K 608..639, zero-pad K >= 620 */
    {
        const int kc = KC - 1, kbase = kc * 32;
        v16bf b[NT];
#pragma unroll
        for (int nt = 0; nt < NT; ++nt)
            b[nt] = *(const v16bf*)(wpk + (((nt * KC + kc) * 32 + lane) << 3));
        v16us au{};
#pragma unroll
        for (int v = 0; v < 8; ++v) {
            int k = kbase + (half << 4) + (v << 1);
            float f0 = 0.f, f1 = 0.f;
            if (k < FEAT) { float2 xy = *(const float2*)(xrow + k); f0 = xy.x; f1 = xy.y; }
            au[2 * v]     = f2bf(f0);
            au[2 * v + 1] = f2bf(f1);
        }
        v16bf a = __builtin_bit_cast(v16bf, au);
#pragma unroll
        for (int nt = 0; nt < NT; ++nt)
            acc[nt] = __builtin_amdgcn_wmma_f32_16x16x32_bf16(
                false, a, false, b[nt], (short)0, acc[nt], false, false);
    }

    /* C layout: lane -> N = lane%16, rows r -> M = r + (lane/16)*8 */
#pragma unroll
    for (int nt = 0; nt < NT; ++nt) {
        int n = nt * 16 + (lane & 15);
        float bias = bh0[n];
        int mb = m0 + (half << 3);
#pragma unroll
        for (int r = 0; r < 8; ++r)
            P[(size_t)(mb + r) * NH + n] = acc[nt][r] + bias;
    }
}

/* ---- kernel 3: sequential recurrence, 32 blocks x 128 threads,
        bf16 weight slices resident in LDS, 3 grid-syncs per step ---- */
__global__ __launch_bounds__(RTH) void rnn_scan(
    const float* __restrict__ P,
    const unsigned* __restrict__ W0s, const unsigned* __restrict__ W1,
    const unsigned* __restrict__ W2,
    const float* __restrict__ bh1, const float* __restrict__ bh2,
    const float* __restrict__ Wc0, const float* __restrict__ bc0,
    const float* __restrict__ Wc1, const float* __restrict__ bc1,
    const int* __restrict__ margin,
    float* gstate, float* gh0, float* gh1,
    unsigned* scnt, unsigned* sgen, float* out)
{
    __shared__ unsigned sW0s[4 * 1088];   /* 4 rows of Wh0s  (17 KB) */
    __shared__ unsigned sW1[4 * 64];      /* 4 rows of Wh1   ( 1 KB) */
    __shared__ unsigned sW2[68 * 64];     /* 68 rows of Wh2  (17 KB) */
    __shared__ float    sState[STATE];    /* 8.5 KB */
    __shared__ float    sVec[NH3];        /* 1 KB   */

    const int tid = threadIdx.x, blk = blockIdx.x;
    const int j = tid >> 5, lane = tid & 31;

    for (int i = tid; i < 4 * 1088; i += RTH) sW0s[i] = W0s[blk * 4 * 1088 + i];
    for (int i = tid; i < 4 * 64;   i += RTH) sW1[i]  = W1[blk * 4 * 64 + i];
    for (int i = tid; i < 68 * 64;  i += RTH) sW2[i]  = W2[blk * 68 * 64 + i];
    const float b1 = bh1[blk * 4 + j];
    __syncthreads();

    for (int t = 0; t < MTOT; ++t) {
        /* ---- phase 1: h0 slice (4 neurons/block, 32 lanes each) ---- */
        for (int i = tid; i < STATE; i += RTH) sState[i] = gstate[i];
        __syncthreads();
        {
            const unsigned* wr = sW0s + j * 1088;
            float acc = 0.f;
            for (int k2 = lane; k2 < 1088; k2 += 32) {
                unsigned u = wr[k2];
                float2 sv = *(const float2*)(sState + 2 * k2);
                acc += bflo(u) * sv.x + bfhi(u) * sv.y;
            }
            acc = wred(acc);
            if (lane == 0) {
                int n = blk * 4 + j;
                gh0[n] = fmaxf(P[(size_t)t * NH + n] + acc, 0.f);
            }
        }
        gsync(scnt, sgen, NBLK);

        /* ---- phase 2: h1 slice ---- */
        if (tid < NH) sVec[tid] = gh0[tid];
        __syncthreads();
        {
            const unsigned* wr = sW1 + j * 64;
            float acc = 0.f;
            for (int k2 = lane; k2 < 64; k2 += 32) {
                unsigned u = wr[k2];
                float2 hv = *(const float2*)(sVec + 2 * k2);
                acc += bflo(u) * hv.x + bfhi(u) * hv.y;
            }
            acc = wred(acc);
            if (lane == 0) gh1[blk * 4 + j] = fmaxf(acc + b1, 0.f);
        }
        gsync(scnt, sgen, NBLK);

        /* ---- phase 3: state slice (68 neurons/block) ---- */
        if (tid < NH) sVec[tid] = gh1[tid];
        __syncthreads();
        if (tid < 68) {
            const unsigned* wr = sW2 + tid * 64;
            float acc = 0.f;
#pragma unroll 8
            for (int k2 = 0; k2 < 64; ++k2) {
                unsigned u = wr[k2];
                float2 hv = *(const float2*)(sVec + 2 * k2);
                acc += bflo(u) * hv.x + bfhi(u) * hv.y;
            }
            int n = blk * 68 + tid;
            gstate[n] = acc + bh2[n];
        }
        gsync(scnt, sgen, NBLK);

        /* ---- per-sequence readout (block 0, fp32) ---- */
        if (((t + 1) & (TSEQ - 1)) == 0 && blk == 0) {
            for (int i = tid; i < STATE; i += RTH) sState[i] = gstate[i];
            __syncthreads();
            for (int n = tid; n < NH3; n += RTH) {
                float acc = bc0[n];
                const float* w = Wc0 + (size_t)n * STATE;
                for (int k = 0; k < STATE; ++k) acc += w[k] * sState[k];
                sVec[n] = fmaxf(acc, 0.f);
            }
            __syncthreads();
            if (tid < 2) {
                float acc = bc1[tid];
                const float* w = Wc1 + tid * NH3;
                for (int k = 0; k < NH3; ++k) acc += w[k] * sVec[k];
                int s = t >> 11;
                out[s * 2 + tid] = tanhf(acc) * (float)(*margin);
            }
            __syncthreads();
        }
    }
}

extern "C" void kernel_launch(void* const* d_in, const int* in_sizes, int n_in,
                              void* d_out, int out_size, void* d_ws, size_t ws_size,
                              hipStream_t stream) {
    const float* fvs = (const float*)d_in[0];
    const float* Wh0 = (const float*)d_in[1];
    const float* bh0 = (const float*)d_in[2];
    const float* Wh1 = (const float*)d_in[3];
    const float* bh1 = (const float*)d_in[4];
    const float* Wh2 = (const float*)d_in[5];
    const float* bh2 = (const float*)d_in[6];
    const float* Wc0 = (const float*)d_in[7];
    const float* bc0 = (const float*)d_in[8];
    const float* Wc1 = (const float*)d_in[9];
    const float* bc1 = (const float*)d_in[10];
    const int* margin = (const int*)d_in[11];
    float* out = (float*)d_out;

    char* ws = (char*)d_ws;
    float*    P      = (float*)(ws + OFF_P);
    unsigned* wpk    = (unsigned*)(ws + OFF_WPK);
    unsigned* w0s    = (unsigned*)(ws + OFF_W0S);
    unsigned* w1     = (unsigned*)(ws + OFF_W1);
    unsigned* w2     = (unsigned*)(ws + OFF_W2);
    float*    gstate = (float*)(ws + OFF_GSTATE);
    float*    gh0    = (float*)(ws + OFF_GH0);
    float*    gh1    = (float*)(ws + OFF_GH1);
    unsigned* sync2  = (unsigned*)(ws + OFF_SYNC);

    rnn_init<<<(STATE + 255) / 256, 256, 0, stream>>>(gstate, sync2);
    rnn_pack<<<1280, 256, 0, stream>>>(Wh0, Wh1, Wh2, wpk, w0s, w1, w2);
    rnn_gemm<<<128, 256, 0, stream>>>(fvs, bh0, wpk, P);
    rnn_scan<<<NBLK, RTH, 0, stream>>>(P, w0s, w1, w2, bh1, bh2,
                                       Wc0, bc0, Wc1, bc1, margin,
                                       gstate, gh0, gh1,
                                       &sync2[0], &sync2[1], out);
}
